// RelativeMultiHeadAttention_3453153706148
// MI455X (gfx1250) — matrix operand order repaired and checked by reading, not compiled, not run on hardware
//
#include <hip/hip_runtime.h>

// ---------------- problem constants ----------------
constexpr int BATCH  = 2;
constexpr int SEQ    = 2048;   // SQ == SK
constexpr int DMODEL = 1024;
constexpr int NHEAD  = 16;
constexpr int HDIM   = 64;
constexpr int NPOS   = 257;    // 2*R+1
constexpr int RELR   = 128;
constexpr int RADIUS = 256;

constexpr size_t N_QKV = (size_t)BATCH * SEQ * DMODEL;  // 4194304
constexpr size_t N_W   = (size_t)DMODEL * DMODEL;       // 1048576
constexpr size_t N_E   = (size_t)NPOS * DMODEL;         // 263168

// workspace offsets (in ushort elements)
constexpr size_t QBF  = 0;
constexpr size_t KBF  = QBF  + N_QKV;
constexpr size_t VBF  = KBF  + N_QKV;
constexpr size_t WQBF = VBF  + N_QKV;
constexpr size_t WKBF = WQBF + N_W;
constexpr size_t WVBF = WKBF + N_W;
constexpr size_t WOBF = WVBF + N_W;
constexpr size_t EBF  = WOBF + N_W;
constexpr size_t QH   = EBF  + N_E;       // [B,H,S,HD] bf16
constexpr size_t KH   = QH   + N_QKV;
constexpr size_t VH   = KH   + N_QKV;
constexpr size_t ATTN = VH   + N_QKV;     // [B,S,D] bf16 (pre-Wo)
// total = ATTN + N_QKV = 33,817,600 ushorts ~ 67.6 MB

// ---------------- WMMA types ----------------
typedef __bf16 bf16_t;
typedef bf16_t v16bf __attribute__((ext_vector_type(16)));
typedef float  v8f   __attribute__((ext_vector_type(8)));

union Frag16 {                 // 16 bf16 = 8 VGPRs = 32 bytes
    v16bf          v;
    uint4          q[2];
    unsigned short u[16];
};

__device__ __forceinline__ v8f wmma_bf16(const v16bf& a, const v16bf& b, v8f c) {
    return __builtin_amdgcn_wmma_f32_16x16x32_bf16(
        /*neg_a=*/false, a, /*neg_b=*/false, b,
        /*c_mod=*/(short)0, c, /*reuse_a=*/false, /*reuse_b=*/false);
}

__device__ __forceinline__ unsigned short f2bf(float f) {
    unsigned int u = __float_as_uint(f);
    u += 0x7fffu + ((u >> 16) & 1u);     // round-to-nearest-even
    return (unsigned short)(u >> 16);
}

static __device__ __forceinline__ int imin(int a, int b) { return a < b ? a : b; }
static __device__ __forceinline__ int imax(int a, int b) { return a > b ? a : b; }

// ---------------- kernel 0: fp32 -> bf16 ----------------
__global__ void cvt_kernel(const float* __restrict__ src,
                           unsigned short* __restrict__ dst, int n) {
    int i = blockIdx.x * blockDim.x + threadIdx.x;
    if (i < n) dst[i] = f2bf(src[i]);
}

// ---------------- kernel 1: fused projection + head split ----------------
// grid: (row_tiles=256, col_groups=8, mat=3), block: 32 (one wave)
// out tile per wave: 16 rows x 128 cols, K = 1024 in steps of 32.
__global__ __launch_bounds__(32) void proj_kernel(const unsigned short* __restrict__ ws,
                                                  unsigned short* __restrict__ wsout) {
    const int lane = threadIdx.x;
    const int half = lane >> 4;
    const int l16  = lane & 15;
    const int rt   = blockIdx.x;     // 16-row tile over B*SEQ
    const int cg   = blockIdx.y;     // 128-col group
    const int mat  = blockIdx.z;     // 0=q 1=k 2=v

    const unsigned short* X = ws + (mat == 0 ? QBF  : mat == 1 ? KBF  : VBF);
    const unsigned short* W = ws + (mat == 0 ? WQBF : mat == 1 ? WKBF : WVBF);
    unsigned short* OUT     = wsout + (mat == 0 ? QH : mat == 1 ? KH : VH);

    v8f acc[8] = {};
    const int arow = rt * 16 + l16;                       // A row for this lane

    for (int kt = 0; kt < DMODEL / 32; ++kt) {
        const int k0 = kt * 32;
        Frag16 a;                                         // 16x32 bf16 A fragment
        a.q[0] = *(const uint4*)(X + (size_t)arow * DMODEL + k0 + half * 16);
        a.q[1] = *(const uint4*)(X + (size_t)arow * DMODEL + k0 + half * 16 + 8);
#pragma unroll
        for (int f = 0; f < 8; ++f) {
            const int ncol = cg * 128 + f * 16 + l16;     // W row == output col
            Frag16 bfr;                                   // 32x16 bf16 B fragment
            const uint4* pb = (const uint4*)(W + (size_t)ncol * DMODEL + k0 + half * 16);
            bfr.q[0] = pb[0];
            bfr.q[1] = pb[1];
            acc[f] = wmma_bf16(a.v, bfr.v, acc[f]);
        }
    }
    // store head-split [B,H,S,HD] bf16
#pragma unroll
    for (int f = 0; f < 8; ++f) {
        const int col = cg * 128 + f * 16 + l16;
        const int h   = col >> 6;
        const int hd  = col & 63;
#pragma unroll
        for (int r = 0; r < 8; ++r) {
            const int m    = half * 8 + r;
            const int grow = rt * 16 + m;
            const int b    = grow >> 11;                  // / SEQ
            const int s    = grow & (SEQ - 1);
            OUT[(((size_t)b * NHEAD + h) * SEQ + s) * HDIM + hd] = f2bf(acc[f][r]);
        }
    }
}

// ---------------- async staging helpers (CDNA5 ASYNCcnt path) ----------------
// Stage one 32-key x 64-dim bf16 tile (4 KB) from global into LDS:
// 8 issues of global_load_async_to_lds_b128 (each lane moves 16 bytes).
__device__ __forceinline__ void async_stage_pair(const unsigned short* __restrict__ src,
                                                 unsigned short* lds_dst,
                                                 int jp, int lane) {
    const unsigned ldsbase = (unsigned)(uintptr_t)lds_dst;   // low 32 bits = LDS offset
#pragma unroll
    for (int c = 0; c < 8; ++c) {
        const int linear = c * 512 + lane * 16;              // byte offset in 4KB tile
        const int row    = linear >> 7;                      // 128 B per key row
        const int colb   = linear & 127;
        const int j      = jp * 16 + row;
        const int jc     = j < SEQ ? j : SEQ - 1;
        const unsigned lds_addr = ldsbase + (unsigned)linear;
        const unsigned long long ga =
            (unsigned long long)(uintptr_t)(src + (size_t)jc * HDIM) + (unsigned)colb;
        asm volatile("global_load_async_to_lds_b128 %0, %1, off"
                     :: "v"(lds_addr), "v"(ga) : "memory");
    }
}

__device__ __forceinline__ void wait_async_pending16() {
    asm volatile("s_wait_asynccnt 0x10" ::: "memory");   // allow 16 in flight (next pair)
}
__device__ __forceinline__ void wait_async_all() {
    asm volatile("s_wait_asynccnt 0x0" ::: "memory");
}

// ---------------- kernel 2: local-window flash attention ----------------
// grid: (q_tiles=128, B*H=32), block: 32 (one wave per 16-row q tile)
__global__ __launch_bounds__(32) void attn_kernel(const unsigned short* __restrict__ ws,
                                                  unsigned short* __restrict__ wsout) {
    __shared__ float qrel[16 * 272];                      // Q_rel tile (padded P)
    __shared__ __align__(16) unsigned short pstage[16 * 32];
    __shared__ __align__(16) unsigned short kstage[2][32 * 64];  // double-buffered K pair tile
    __shared__ __align__(16) unsigned short vstage[2][32 * 64];  // double-buffered V pair tile

    const int lane = threadIdx.x;
    const int half = lane >> 4;
    const int l16  = lane & 15;
    const int qt   = blockIdx.x;
    const int bh   = blockIdx.y;                          // b*16 + h
    const int b    = bh >> 4;
    const int h    = bh & 15;

    const unsigned short* qh  = ws + QH + (size_t)bh * SEQ * HDIM;
    const unsigned short* kh  = ws + KH + (size_t)bh * SEQ * HDIM;
    const unsigned short* vh  = ws + VH + (size_t)bh * SEQ * HDIM;
    const unsigned short* Eb  = ws + EBF;
    unsigned short* attnb     = wsout + ATTN;

    const int i0 = qt * 16;

    const int jlo = imax(0, i0 - RADIUS) >> 4;
    const int jhi = imin(SEQ - 1, i0 + 15 + RADIUS) >> 4;

    // kick off async staging of the first key/value pair tile
    async_stage_pair(kh, &kstage[0][0], jlo, lane);
    async_stage_pair(vh, &vstage[0][0], jlo, lane);

    // ---- load q A-fragments (16x64 -> two 16x32 frags), kept in registers ----
    Frag16 qa[2];
    {
        const int r = i0 + l16;
#pragma unroll
        for (int kt = 0; kt < 2; ++kt) {
            qa[kt].q[0] = *(const uint4*)(qh + (size_t)r * HDIM + kt * 32 + half * 16);
            qa[kt].q[1] = *(const uint4*)(qh + (size_t)r * HDIM + kt * 32 + half * 16 + 8);
        }
    }

    // ---- Q_rel tile (16 x 257, padded to 272) via WMMA into LDS ----
    for (int pt = 0; pt < 17; ++pt) {
        v8f c = {};
#pragma unroll
        for (int kt = 0; kt < 2; ++kt) {
            Frag16 rb;                                    // B[k=d, n=p] = E[p, d*16+h]
            const int p = pt * 16 + l16;
#pragma unroll
            for (int dd = 0; dd < 16; ++dd) {
                const int d = kt * 32 + half * 16 + dd;
                rb.u[dd] = (p < NPOS) ? Eb[(size_t)p * DMODEL + d * NHEAD + h]
                                      : (unsigned short)0;
            }
            c = wmma_bf16(qa[kt].v, rb.v, c);
        }
#pragma unroll
        for (int r = 0; r < 8; ++r)
            qrel[(half * 8 + r) * 272 + pt * 16 + l16] = c[r];
    }
    __syncthreads();

    // ---- online-softmax loop over local key window ----
    v8f acc[4] = {};
    float mrow[8], lrow[8];
#pragma unroll
    for (int r = 0; r < 8; ++r) { mrow[r] = -INFINITY; lrow[r] = 0.f; }

    int buf = 0;
    for (int jp = jlo; jp <= jhi; jp += 2, buf ^= 1) {
        // issue the next pair's async staging before consuming the current one
        const bool have_next = (jp + 2 <= jhi);
        if (have_next) {
            async_stage_pair(kh, &kstage[buf ^ 1][0], jp + 2, lane);
            async_stage_pair(vh, &vstage[buf ^ 1][0], jp + 2, lane);
            wait_async_pending16();                      // current 16 done, next in flight
        } else {
            wait_async_all();
        }
        const unsigned short* kst = &kstage[buf][0];
        const unsigned short* vst = &vstage[buf][0];

        // hint the K rows two pairs ahead toward the WGP cache
        __builtin_prefetch(kh + (size_t)imin((jp + 4) * 16, SEQ - 1) * HDIM, 0, 1);

        // S = q . k^T for two 16-key tiles (K frags via ds_load_b128 from LDS)
        v8f s[2] = {};
#pragma unroll
        for (int t = 0; t < 2; ++t) {
            const int jloc = t * 16 + l16;               // local key row in pair tile
#pragma unroll
            for (int kt = 0; kt < 2; ++kt) {
                Frag16 kb;
                const uint4* pk =
                    (const uint4*)(kst + (size_t)jloc * HDIM + kt * 32 + half * 16);
                kb.q[0] = pk[0];
                kb.q[1] = pk[1];
                s[t] = wmma_bf16(qa[kt].v, kb.v, s[t]);
            }
        }

        // rel-shift add, scale, window mask, online softmax update
        float fac[8];
#pragma unroll
        for (int r = 0; r < 8; ++r) {
            const int m = half * 8 + r;
            const int i = i0 + m;
            float mx = -INFINITY;
#pragma unroll
            for (int t = 0; t < 2; ++t) {
                const int j   = (jp + t) * 16 + l16;
                const int dd  = j - i;
                const int rid = imin(imax(dd, -RELR), RELR) + RELR;
                float val = (s[t][r] + qrel[m * 272 + rid]) * 0.125f; // 1/sqrt(64)
                const bool masked = (j >= SEQ) | (dd > RADIUS) | (dd < -RADIUS);
                val = masked ? -INFINITY : val;
                s[t][r] = val;
                mx = fmaxf(mx, val);
            }
#pragma unroll
            for (int off = 1; off < 16; off <<= 1)
                mx = fmaxf(mx, __shfl_xor(mx, off, 32));
            const float mn    = fmaxf(mrow[r], mx);
            const float msafe = (mn == -INFINITY) ? 0.f : mn;
            const float f     = (mrow[r] == -INFINITY) ? 0.f : __expf(mrow[r] - mn);
            float psum = 0.f;
#pragma unroll
            for (int t = 0; t < 2; ++t) {
                const float pv = __expf(s[t][r] - msafe);
                s[t][r] = pv;
                psum += pv;
            }
#pragma unroll
            for (int off = 1; off < 16; off <<= 1)
                psum += __shfl_xor(psum, off, 32);
            lrow[r] = lrow[r] * f + psum;
            mrow[r] = mn;
            fac[r]  = f;
        }
#pragma unroll
        for (int ff = 0; ff < 4; ++ff)
#pragma unroll
            for (int r = 0; r < 8; ++r) acc[ff][r] *= fac[r];

        // stage P (D-frag layout) -> LDS -> reload as 16x32 A fragment
#pragma unroll
        for (int t = 0; t < 2; ++t)
#pragma unroll
            for (int r = 0; r < 8; ++r)
                pstage[(half * 8 + r) * 32 + t * 16 + l16] = f2bf(s[t][r]);
        __syncthreads();
        Frag16 pa;
        pa.q[0] = *(const uint4*)(pstage + l16 * 32 + half * 16);
        pa.q[1] = *(const uint4*)(pstage + l16 * 32 + half * 16 + 8);
        __syncthreads();

        // acc += P . V   (V B-fragments gathered from LDS: B[k=key, n=hd])
#pragma unroll
        for (int ff = 0; ff < 4; ++ff) {
            Frag16 vb;
#pragma unroll
            for (int kk = 0; kk < 16; ++kk) {
                const int jloc = half * 16 + kk;
                vb.u[kk] = vst[(size_t)jloc * HDIM + ff * 16 + l16];
            }
            acc[ff] = wmma_bf16(pa.v, vb.v, acc[ff]);
        }
    }

    // ---- epilogue: normalize, store as [B, S, hd*H + h] bf16 ----
#pragma unroll
    for (int ff = 0; ff < 4; ++ff) {
        const int hd = ff * 16 + l16;
#pragma unroll
        for (int r = 0; r < 8; ++r) {
            const int m  = half * 8 + r;
            const int si = i0 + m;
            const float o = acc[ff][r] / lrow[r];
            attnb[((size_t)b * SEQ + si) * DMODEL + hd * NHEAD + h] = f2bf(o);
        }
    }
}

// ---------------- kernel 3: output projection (fp32 out) ----------------
// grid: (row_tiles=256, col_groups=8), block: 32
__global__ __launch_bounds__(32) void outproj_kernel(const unsigned short* __restrict__ ws,
                                                     float* __restrict__ out) {
    const int lane = threadIdx.x;
    const int half = lane >> 4;
    const int l16  = lane & 15;
    const int rt   = blockIdx.x;
    const int cg   = blockIdx.y;

    const unsigned short* X = ws + ATTN;
    const unsigned short* W = ws + WOBF;

    v8f acc[8] = {};
    const int arow = rt * 16 + l16;

    for (int kt = 0; kt < DMODEL / 32; ++kt) {
        const int k0 = kt * 32;
        Frag16 a;
        a.q[0] = *(const uint4*)(X + (size_t)arow * DMODEL + k0 + half * 16);
        a.q[1] = *(const uint4*)(X + (size_t)arow * DMODEL + k0 + half * 16 + 8);
#pragma unroll
        for (int f = 0; f < 8; ++f) {
            const int ncol = cg * 128 + f * 16 + l16;
            Frag16 bfr;
            const uint4* pb = (const uint4*)(W + (size_t)ncol * DMODEL + k0 + half * 16);
            bfr.q[0] = pb[0];
            bfr.q[1] = pb[1];
            acc[f] = wmma_bf16(a.v, bfr.v, acc[f]);
        }
    }
#pragma unroll
    for (int f = 0; f < 8; ++f) {
        const int col = cg * 128 + f * 16 + l16;
#pragma unroll
        for (int r = 0; r < 8; ++r) {
            const int grow = rt * 16 + half * 8 + r;
            out[(size_t)grow * DMODEL + col] = acc[f][r];
        }
    }
}

// ---------------- host launcher ----------------
extern "C" void kernel_launch(void* const* d_in, const int* in_sizes, int n_in,
                              void* d_out, int out_size, void* d_ws, size_t ws_size,
                              hipStream_t stream) {
    (void)in_sizes; (void)n_in; (void)out_size; (void)ws_size;
    const float* Q  = (const float*)d_in[0];
    const float* K  = (const float*)d_in[1];
    const float* V  = (const float*)d_in[2];
    const float* Wq = (const float*)d_in[3];
    const float* Wk = (const float*)d_in[4];
    const float* Wv = (const float*)d_in[5];
    const float* Wo = (const float*)d_in[6];
    const float* E  = (const float*)d_in[7];

    unsigned short* ws = (unsigned short*)d_ws;

    auto cvt = [&](const float* src, size_t off, size_t n) {
        const int blocks = (int)((n + 255) / 256);
        cvt_kernel<<<blocks, 256, 0, stream>>>(src, ws + off, (int)n);
    };
    cvt(Q,  QBF,  N_QKV);
    cvt(K,  KBF,  N_QKV);
    cvt(V,  VBF,  N_QKV);
    cvt(Wq, WQBF, N_W);
    cvt(Wk, WKBF, N_W);
    cvt(Wv, WVBF, N_W);
    cvt(Wo, WOBF, N_W);
    cvt(E,  EBF,  N_E);

    dim3 gA((BATCH * SEQ) / 16, DMODEL / 128, 3);
    proj_kernel<<<gA, 32, 0, stream>>>(ws, ws);

    dim3 gB(SEQ / 16, BATCH * NHEAD);
    attn_kernel<<<gB, 32, 0, stream>>>(ws, ws);

    dim3 gC((BATCH * SEQ) / 16, DMODEL / 128);
    outproj_kernel<<<gC, 32, 0, stream>>>(ws, (float*)d_out);
}
